// TransMIL_73873437491685
// MI455X (gfx1250) — matrix operand order repaired and checked by reading, not compiled, not run on hardware
//
#include <hip/hip_runtime.h>
#include <hip/hip_bf16.h>

// ---------------- constants ----------------
#define NTOK   8192
#define INDIM  1024
#define DMODEL 512
#define NH     8
#define DH     64
#define MLM    256     // landmarks
#define LFAC   33      // Np / MLM
#define NP     8448    // padded sequence for attention (128*66)
#define NTOT   8282    // 1 cls + 8281 tokens
#define PADR   166     // front zero-pad rows inside attention
#define NIMG   8281
#define HS     91
#define ADD_DUP 89
#define SCALE_Q 0.125f // DH^-0.5
#define KRES   33

#define GRID1(n) ((int)((((long long)(n)) + 255) / 256))

typedef __attribute__((ext_vector_type(16))) __bf16 v16bf;
typedef __attribute__((ext_vector_type(8)))  __bf16 v8bf;
typedef __attribute__((ext_vector_type(8)))  float  v8f;

// ---------------- batched WMMA GEMM: C = act(A @ B(^T) + bias) ----------------
// REQUIREMENTS (all call sites satisfy them): M % 128 == 0, N % 64 == 0,
// K % 64 == 0, and every matrix has < 2^31 elements (32-bit offsets).
// Block tile 128(M) x 64(N) x 64(K); 8 waves in 4x2 grid; each wave owns a
// 32x32 C patch = 4 WMMA accumulators; 8 v_wmma_f32_16x16x32_bf16 per stage.
// Double-buffered LDS: stage k+1 overlaps WMMA compute of stage k.
#define BM 128
#define BN 64
#define BKK 64

__device__ __forceinline__ void stage_tiles(
    const float* __restrict__ A, const float* __restrict__ B,
    int Kdim, int Ndim, int transB, int m0, int n0, int k0,
    int lrow, int lcol, __bf16 (*As)[BKK], __bf16 (*Bs)[BKK])
{
  // batch all global loads first (no per-load waits), then convert + LDS-store
  float4 av[8];
#pragma unroll
  for (int i = 0; i < 8; ++i)
    av[i] = *(const float4*)(A + (unsigned)((m0 + lrow + i * 16) * Kdim + k0 + lcol));
  float4 bv[4];
  if (transB) {        // B is N x K
#pragma unroll
    for (int i = 0; i < 4; ++i)
      bv[i] = *(const float4*)(B + (unsigned)((n0 + lrow + i * 16) * Kdim + k0 + lcol));
  } else {             // B is K x N
#pragma unroll
    for (int i = 0; i < 4; ++i)
      bv[i] = *(const float4*)(B + (unsigned)((k0 + lrow + i * 16) * Ndim + n0 + lcol));
  }
#pragma unroll
  for (int i = 0; i < 8; ++i) {
    int r = lrow + i * 16;
    As[r][lcol + 0] = (__bf16)av[i].x;
    As[r][lcol + 1] = (__bf16)av[i].y;
    As[r][lcol + 2] = (__bf16)av[i].z;
    As[r][lcol + 3] = (__bf16)av[i].w;
  }
  if (transB) {
#pragma unroll
    for (int i = 0; i < 4; ++i) {
      int r = lrow + i * 16;     // n-local
      Bs[r][lcol + 0] = (__bf16)bv[i].x;
      Bs[r][lcol + 1] = (__bf16)bv[i].y;
      Bs[r][lcol + 2] = (__bf16)bv[i].z;
      Bs[r][lcol + 3] = (__bf16)bv[i].w;
    }
  } else {
#pragma unroll
    for (int i = 0; i < 4; ++i) {
      int r = lrow + i * 16;     // k-local; scatter transpose into Bs[n][k]
      Bs[lcol + 0][r] = (__bf16)bv[i].x;
      Bs[lcol + 1][r] = (__bf16)bv[i].y;
      Bs[lcol + 2][r] = (__bf16)bv[i].z;
      Bs[lcol + 3][r] = (__bf16)bv[i].w;
    }
  }
}

__global__ void __launch_bounds__(256)
wmma_gemm_kernel(const float* __restrict__ A, const float* __restrict__ B,
                 const float* __restrict__ bias, float* __restrict__ C,
                 int Mdim, int Ndim, int Kdim,
                 long long sA, long long sB, long long sC,
                 int transB, int relu)
{
  A += (long long)blockIdx.z * sA;
  B += (long long)blockIdx.z * sB;
  C += (long long)blockIdx.z * sC;

  __shared__ __bf16 As[2][BM][BKK];   // 2 x 16 KB
  __shared__ __bf16 Bs[2][BN][BKK];   // 2 x  8 KB

  const int tid  = threadIdx.x;
  const int lane = tid & 31;
  const int wid  = tid >> 5;
  const int wr   = wid >> 1;       // 0..3 (M direction)
  const int wc   = wid & 1;        // 0..1 (N direction)
  const int m0   = blockIdx.y * BM;
  const int n0   = blockIdx.x * BN;

  const int lrow = tid >> 4;       // 0..15, loader row
  const int lcol = (tid & 15) * 4; // 0..60, loader col (float4 granularity)
  const int lml  = lane & 15;
  const int kbA  = (lane >> 4) * 8;   // A frag K base within 32-step
  const int kbB  = (lane >> 4) * 8;   // B frag K base within 32-step

  v8f acc[2][2] = {};

  // prologue: stage first tile into buffer 0
  stage_tiles(A, B, Kdim, Ndim, transB, m0, n0, 0, lrow, lcol, As[0], Bs[0]);
  __syncthreads();

  int buf = 0;
  for (int k0 = 0; k0 < Kdim; k0 += BKK) {
    // stage next tile into the alternate buffer while computing this one
    int nk = k0 + BKK;
    if (nk < Kdim)
      stage_tiles(A, B, Kdim, Ndim, transB, m0, n0, nk, lrow, lcol,
                  As[buf ^ 1], Bs[buf ^ 1]);

    __bf16 (*Ac)[BKK] = As[buf];
    __bf16 (*Bc)[BKK] = Bs[buf];
#pragma unroll
    for (int kk = 0; kk < BKK; kk += 32) {
      v16bf af[2], bfm[2];
#pragma unroll
      for (int mi = 0; mi < 2; ++mi) {
        const __bf16* pa = &Ac[wr * 32 + mi * 16 + lml][kk + kbA];
        v8bf lo = *(const v8bf*)pa;          // K = kb .. kb+7
        v8bf hi = *(const v8bf*)(pa + 16);   // K = kb+16 .. kb+23
        af[mi] = __builtin_shufflevector(lo, hi, 0, 1, 2, 3, 4, 5, 6, 7,
                                                 8, 9, 10, 11, 12, 13, 14, 15);
      }
#pragma unroll
      for (int ni = 0; ni < 2; ++ni) {
        const __bf16* pb = &Bc[wc * 32 + ni * 16 + lml][kk + kbB];
        v8bf lo = *(const v8bf*)pb;          // K = kb .. kb+7
        v8bf hi = *(const v8bf*)(pb + 16);   // K = kb+16 .. kb+23
        bfm[ni] = __builtin_shufflevector(lo, hi, 0, 1, 2, 3, 4, 5, 6, 7,
                                                  8, 9, 10, 11, 12, 13, 14, 15);
      }
#pragma unroll
      for (int mi = 0; mi < 2; ++mi)
#pragma unroll
        for (int ni = 0; ni < 2; ++ni)
          acc[mi][ni] = __builtin_amdgcn_wmma_f32_16x16x32_bf16(
              false, af[mi], false, bfm[ni], (short)0, acc[mi][ni], false, false);
    }
    __syncthreads();
    buf ^= 1;
  }

  // ---- store C 32x32 per wave ----
#pragma unroll
  for (int ni = 0; ni < 2; ++ni) {
    int gn = n0 + wc * 32 + ni * 16 + lml;
    float bv = (bias != nullptr) ? bias[gn] : 0.f;
#pragma unroll
    for (int mi = 0; mi < 2; ++mi) {
#pragma unroll
      for (int r = 0; r < 8; ++r) {
        int gm = m0 + wr * 32 + mi * 16 + r + ((lane >> 4) << 3);
        float v = acc[mi][ni][r] + bv;
        if (relu) v = fmaxf(v, 0.f);
        C[(unsigned)(gm * Ndim + gn)] = v;
      }
    }
  }
}

// ---------------- elementwise / reduction kernels ----------------
__global__ void fill_zero(float* p, long long n) {
  long long i = blockIdx.x * 256LL + threadIdx.x;
  if (i < n) p[i] = 0.f;
}
__global__ void copy_buf(const float* a, float* o, long long n) {
  long long i = blockIdx.x * 256LL + threadIdx.x;
  if (i < n) o[i] = a[i];
}
__global__ void finish_embed(float* h, const float* cls) {
  int i = blockIdx.x * 256 + threadIdx.x;
  if (i < DMODEL) h[i] = cls[i];
  if (i < ADD_DUP * DMODEL)
    h[(long long)(1 + NTOK) * DMODEL + i] = h[DMODEL + i];
}
__global__ void layernorm_rows(const float* __restrict__ in, float* __restrict__ out,
                               const float* __restrict__ g, const float* __restrict__ b,
                               int rowOff) {
  int row = blockIdx.x, t = threadIdx.x;
  const float* p = in + (long long)row * DMODEL;
  float* q = out + (long long)(row + rowOff) * DMODEL;
  __shared__ float red[256];
  float a0 = p[t], a1 = p[t + 256];
  red[t] = a0 + a1; __syncthreads();
  for (int s = 128; s > 0; s >>= 1) { if (t < s) red[t] += red[t + s]; __syncthreads(); }
  float mu = red[0] * (1.f / 512.f); __syncthreads();
  float d0 = a0 - mu, d1 = a1 - mu;
  red[t] = d0 * d0 + d1 * d1; __syncthreads();
  for (int s = 128; s > 0; s >>= 1) { if (t < s) red[t] += red[t + s]; __syncthreads(); }
  float inv = rsqrtf(red[0] * (1.f / 512.f) + 1e-5f);
  q[t]       = d0 * inv * g[t]       + b[t];
  q[t + 256] = d1 * inv * g[t + 256] + b[t + 256];
}
__global__ void split_qkv(const float* __restrict__ qkv, float* __restrict__ q,
                          float* __restrict__ k, float* __restrict__ v) {
  long long i = blockIdx.x * 256LL + threadIdx.x;
  if (i >= (long long)NH * NP * DH) return;
  int d = (int)(i & 63);
  long long t = i >> 6;
  int n = (int)(t % NP);
  int h = (int)(t / NP);
  long long src = (long long)n * (3 * DMODEL) + h * DH + d;
  q[i] = qkv[src] * SCALE_Q;
  k[i] = qkv[src + DMODEL];
  v[i] = qkv[src + 2 * DMODEL];
}
__global__ void landmark_mean(const float* __restrict__ x, float* __restrict__ o) {
  long long i = blockIdx.x * 256LL + threadIdx.x;
  if (i >= (long long)NH * MLM * DH) return;
  int d = (int)(i & 63);
  long long t = i >> 6;
  int m = (int)(t % MLM);
  int h = (int)(t / MLM);
  const float* p = x + ((long long)h * NP + (long long)m * LFAC) * DH + d;
  float s = 0.f;
#pragma unroll
  for (int j = 0; j < LFAC; ++j) s += p[(long long)j * DH];
  o[i] = s * (1.f / (float)LFAC);
}
__global__ void softmax_rows(float* __restrict__ data, int cols) {
  long long row = blockIdx.x;
  float* p = data + row * (long long)cols;
  __shared__ float red[256];
  int t = threadIdx.x;
  float m = -1e30f;
  for (int j = t; j < cols; j += 256) m = fmaxf(m, p[j]);
  red[t] = m; __syncthreads();
  for (int s = 128; s > 0; s >>= 1) { if (t < s) red[t] = fmaxf(red[t], red[t + s]); __syncthreads(); }
  float mx = red[0]; __syncthreads();
  float sum = 0.f;
  for (int j = t; j < cols; j += 256) { float e = __expf(p[j] - mx); p[j] = e; sum += e; }
  red[t] = sum; __syncthreads();
  for (int s = 128; s > 0; s >>= 1) { if (t < s) red[t] += red[t + s]; __syncthreads(); }
  float inv = 1.f / red[0];
  for (int j = t; j < cols; j += 256) p[j] *= inv;
}
__global__ void pinv_sums(const float* __restrict__ a2, float* rsum, float* csum) {
  int bi = blockIdx.x, h = bi >> 8, i = bi & 255, j = threadIdx.x;
  const float* x = a2 + ((long long)h << 16);
  __shared__ float red[256];
  red[j] = fabsf(x[i * 256 + j]); __syncthreads();
  for (int s = 128; s > 0; s >>= 1) { if (j < s) red[j] += red[j + s]; __syncthreads(); }
  if (j == 0) rsum[bi] = red[0];
  __syncthreads();
  red[j] = fabsf(x[j * 256 + i]); __syncthreads();
  for (int s = 128; s > 0; s >>= 1) { if (j < s) red[j] += red[j + s]; __syncthreads(); }
  if (j == 0) csum[bi] = red[0];
}
__global__ void pinv_scalars(const float* rsum, const float* csum, float* scal) {
  int t = threadIdx.x;
  __shared__ float red[256];
  float m = 0.f;
  for (int j = t; j < NH * MLM; j += 256) m = fmaxf(m, rsum[j]);
  red[t] = m; __syncthreads();
  for (int s = 128; s > 0; s >>= 1) { if (t < s) red[t] = fmaxf(red[t], red[t + s]); __syncthreads(); }
  if (t == 0) scal[0] = red[0];
  __syncthreads();
  m = 0.f;
  for (int j = t; j < NH * MLM; j += 256) m = fmaxf(m, csum[j]);
  red[t] = m; __syncthreads();
  for (int s = 128; s > 0; s >>= 1) { if (t < s) red[t] = fmaxf(red[t], red[t + s]); __syncthreads(); }
  if (t == 0) scal[1] = red[0];
}
__global__ void pinv_init_z(const float* __restrict__ a2, const float* scal, float* z) {
  long long i = blockIdx.x * 256LL + threadIdx.x;
  if (i >= (long long)NH * MLM * MLM) return;
  int j  = (int)(i & 255);
  int ii = (int)((i >> 8) & 255);
  int h  = (int)(i >> 16);
  z[i] = a2[((long long)h << 16) + (long long)j * 256 + ii] / (scal[0] * scal[1]);
}
__global__ void diag_sub(const float* __restrict__ in, float* __restrict__ out, float alpha) {
  long long i = blockIdx.x * 256LL + threadIdx.x;
  if (i >= (long long)NH * MLM * MLM) return;
  int j  = (int)(i & 255);
  int ii = (int)((i >> 8) & 255);
  out[i] = ((ii == j) ? alpha : 0.f) - in[i];
}
__global__ void scale_copy(const float* __restrict__ in, float* __restrict__ out, float s) {
  long long i = blockIdx.x * 256LL + threadIdx.x;
  if (i >= (long long)NH * MLM * MLM) return;
  out[i] = s * in[i];
}
__global__ void conv1d_add(float* __restrict__ o, const float* __restrict__ v,
                           const float* __restrict__ w) {
  long long i = blockIdx.x * 256LL + threadIdx.x;
  if (i >= (long long)NH * NP * DH) return;
  int d = (int)(i & 63);
  long long t = i >> 6;
  int n = (int)(t % NP);
  int h = (int)(t / NP);
  const float* wp = w + h * KRES;
  const float* vp = v + (long long)h * NP * DH + d;
  float s = 0.f;
#pragma unroll
  for (int j = 0; j < KRES; ++j) {
    int nn = n + j - (KRES / 2);
    if ((unsigned)nn < (unsigned)NP) s += wp[j] * vp[(long long)nn * DH];
  }
  o[i] += s;
}
__global__ void merge_heads(const float* __restrict__ o, float* __restrict__ m) {
  long long i = blockIdx.x * 256LL + threadIdx.x;
  if (i >= (long long)NH * NP * DH) return;
  int d = (int)(i & 63);
  long long t = i >> 6;
  int n = (int)(t % NP);
  int h = (int)(t / NP);
  m[(long long)n * DMODEL + h * DH + d] = o[i];
}
__global__ void add_resid(float* __restrict__ h, const float* __restrict__ proj) {
  long long i = blockIdx.x * 256LL + threadIdx.x;
  if (i >= (long long)NTOT * DMODEL) return;
  h[i] += proj[(long long)PADR * DMODEL + i];
}
__global__ void ppeg_kernel(const float* __restrict__ h, float* __restrict__ o,
                            const float* __restrict__ w7, const float* __restrict__ b7,
                            const float* __restrict__ w5, const float* __restrict__ b5,
                            const float* __restrict__ w3, const float* __restrict__ b3) {
  long long i = blockIdx.x * 256LL + threadIdx.x;
  if (i >= (long long)NIMG * DMODEL) return;
  int c = (int)(i % DMODEL);
  int p = (int)(i / DMODEL);
  int y = p / HS, x = p % HS;
  float acc = h[(long long)(1 + p) * DMODEL + c];
  const float* w = w7 + c * 49;
  float s = b7[c];
  for (int ky = 0; ky < 7; ++ky) {
    int yy = y + ky - 3; if ((unsigned)yy >= HS) continue;
    for (int kx = 0; kx < 7; ++kx) {
      int xx = x + kx - 3; if ((unsigned)xx >= HS) continue;
      s += w[ky * 7 + kx] * h[(long long)(1 + yy * HS + xx) * DMODEL + c];
    }
  }
  acc += s;
  w = w5 + c * 25; s = b5[c];
  for (int ky = 0; ky < 5; ++ky) {
    int yy = y + ky - 2; if ((unsigned)yy >= HS) continue;
    for (int kx = 0; kx < 5; ++kx) {
      int xx = x + kx - 2; if ((unsigned)xx >= HS) continue;
      s += w[ky * 5 + kx] * h[(long long)(1 + yy * HS + xx) * DMODEL + c];
    }
  }
  acc += s;
  w = w3 + c * 9; s = b3[c];
  for (int ky = 0; ky < 3; ++ky) {
    int yy = y + ky - 1; if ((unsigned)yy >= HS) continue;
    for (int kx = 0; kx < 3; ++kx) {
      int xx = x + kx - 1; if ((unsigned)xx >= HS) continue;
      s += w[ky * 3 + kx] * h[(long long)(1 + yy * HS + xx) * DMODEL + c];
    }
  }
  acc += s;
  o[(long long)(1 + p) * DMODEL + c] = acc;
}
__global__ void final_head(const float* __restrict__ h, const float* __restrict__ g,
                           const float* __restrict__ b, const float* __restrict__ w,
                           const float* __restrict__ bias, float* __restrict__ out) {
  __shared__ float red[512];
  __shared__ float emb[512];
  __shared__ float lg[2];
  int t = threadIdx.x;
  float xv = h[t];
  red[t] = xv; __syncthreads();
  for (int s = 256; s > 0; s >>= 1) { if (t < s) red[t] += red[t + s]; __syncthreads(); }
  float mu = red[0] * (1.f / 512.f); __syncthreads();
  float d = xv - mu;
  red[t] = d * d; __syncthreads();
  for (int s = 256; s > 0; s >>= 1) { if (t < s) red[t] += red[t + s]; __syncthreads(); }
  float inv = rsqrtf(red[0] * (1.f / 512.f) + 1e-5f);
  float e = d * inv * g[t] + b[t];
  emb[t] = e;
  out[5 + t] = e;
  __syncthreads();
  if (t < 2) {
    float s = bias[t];
    for (int j = 0; j < 512; ++j) s += emb[j] * w[j * 2 + t];
    lg[t] = s; out[t] = s;
  }
  __syncthreads();
  if (t == 0) {
    float mx = fmaxf(lg[0], lg[1]);
    float e0 = __expf(lg[0] - mx), e1 = __expf(lg[1] - mx);
    float den = e0 + e1;
    out[2] = e0 / den; out[3] = e1 / den;
    out[4] = (lg[1] > lg[0]) ? 1.f : 0.f;
  }
}

// ---------------- host orchestration ----------------
struct Scratch {
  float *h, *t0, *t1, *qkv, *qh, *kh, *vh, *oh;
  float *ql, *kl, *a2, *z, *xz, *w1, *w2, *a3v;
  float *scal, *rsum, *csum, *big0, *big1;
};

static inline void gemm(hipStream_t st, const float* A, const float* B, const float* bias,
                        float* C, int M, int N, int K,
                        long long sA, long long sB, long long sC, int batch,
                        bool transB, bool relu) {
  dim3 grid((N + BN - 1) / BN, (M + BM - 1) / BM, batch);
  wmma_gemm_kernel<<<grid, 256, 0, st>>>(A, B, bias, C, M, N, K, sA, sB, sC,
                                         (int)transB, (int)relu);
}

static void run_attn(hipStream_t st, float* h, Scratch& S,
                     const float* lng, const float* lnb,
                     const float* qkvw, const float* outw, const float* outb,
                     const float* convw) {
  const long long HSZ = (long long)NH * NP * DH;
  const long long ZSZ = (long long)NH * MLM * MLM;
  // ln(h) into zero-front-padded buffer (rows 0..165 = 0)
  fill_zero<<<GRID1(PADR * DMODEL), 256, 0, st>>>(S.t1, (long long)PADR * DMODEL);
  layernorm_rows<<<NTOT, 256, 0, st>>>(h, S.t1, lng, lnb, PADR);
  // qkv projection
  gemm(st, S.t1, qkvw, nullptr, S.qkv, NP, 3 * DMODEL, DMODEL, 0, 0, 0, 1, false, false);
  split_qkv<<<GRID1(HSZ), 256, 0, st>>>(S.qkv, S.qh, S.kh, S.vh);
  landmark_mean<<<GRID1(NH * MLM * DH), 256, 0, st>>>(S.qh, S.ql);
  landmark_mean<<<GRID1(NH * MLM * DH), 256, 0, st>>>(S.kh, S.kl);
  // a1 = softmax(q @ k_l^T)
  gemm(st, S.qh, S.kl, nullptr, S.big0, NP, MLM, DH,
       (long long)NP * DH, (long long)MLM * DH, (long long)NP * MLM, NH, true, false);
  softmax_rows<<<NH * NP, 256, 0, st>>>(S.big0, MLM);
  // a2 = softmax(q_l @ k_l^T)
  gemm(st, S.ql, S.kl, nullptr, S.a2, MLM, MLM, DH,
       (long long)MLM * DH, (long long)MLM * DH, (long long)MLM * MLM, NH, true, false);
  softmax_rows<<<NH * MLM, 256, 0, st>>>(S.a2, MLM);
  // a3 = softmax(q_l @ k^T)
  gemm(st, S.ql, S.kh, nullptr, S.big1, MLM, NP, DH,
       (long long)MLM * DH, (long long)NP * DH, (long long)MLM * NP, NH, true, false);
  softmax_rows<<<NH * MLM, 256, 0, st>>>(S.big1, NP);
  // a3v = a3 @ v
  gemm(st, S.big1, S.vh, nullptr, S.a3v, MLM, DH, NP,
       (long long)MLM * NP, (long long)NP * DH, (long long)MLM * DH, NH, false, false);
  // Moore-Penrose pinv of a2 (Newton-Schulz, 6 iterations)
  pinv_sums<<<NH * MLM, 256, 0, st>>>(S.a2, S.rsum, S.csum);
  pinv_scalars<<<1, 256, 0, st>>>(S.rsum, S.csum, S.scal);
  pinv_init_z<<<GRID1(ZSZ), 256, 0, st>>>(S.a2, S.scal, S.z);
  for (int it = 0; it < 6; ++it) {
    gemm(st, S.a2, S.z, nullptr, S.xz, MLM, MLM, MLM, ZSZ / NH, ZSZ / NH, ZSZ / NH, NH, false, false);
    diag_sub<<<GRID1(ZSZ), 256, 0, st>>>(S.xz, S.w1, 7.f);
    gemm(st, S.xz, S.w1, nullptr, S.w2, MLM, MLM, MLM, ZSZ / NH, ZSZ / NH, ZSZ / NH, NH, false, false);
    diag_sub<<<GRID1(ZSZ), 256, 0, st>>>(S.w2, S.w1, 15.f);
    gemm(st, S.xz, S.w1, nullptr, S.w2, MLM, MLM, MLM, ZSZ / NH, ZSZ / NH, ZSZ / NH, NH, false, false);
    diag_sub<<<GRID1(ZSZ), 256, 0, st>>>(S.w2, S.w1, 13.f);
    gemm(st, S.z, S.w1, nullptr, S.w2, MLM, MLM, MLM, ZSZ / NH, ZSZ / NH, ZSZ / NH, NH, false, false);
    scale_copy<<<GRID1(ZSZ), 256, 0, st>>>(S.w2, S.z, 0.25f);
  }
  // out = (a1 @ pinv) @ a3v
  gemm(st, S.big0, S.z, nullptr, S.big1, NP, MLM, MLM,
       (long long)NP * MLM, ZSZ / NH, (long long)NP * MLM, NH, false, false);
  gemm(st, S.big1, S.a3v, nullptr, S.oh, NP, DH, MLM,
       (long long)NP * MLM, (long long)MLM * DH, (long long)NP * DH, NH, false, false);
  // + depthwise 1D conv on v, merge heads, output projection, residual
  conv1d_add<<<GRID1(HSZ), 256, 0, st>>>(S.oh, S.vh, convw);
  merge_heads<<<GRID1(HSZ), 256, 0, st>>>(S.oh, S.t1);
  gemm(st, S.t1, outw, outb, S.t0, NP, DMODEL, DMODEL, 0, 0, 0, 1, false, false);
  add_resid<<<GRID1((long long)NTOT * DMODEL), 256, 0, st>>>(h, S.t0);
}

extern "C" void kernel_launch(void* const* d_in, const int* in_sizes, int n_in,
                              void* d_out, int out_size, void* d_ws, size_t ws_size,
                              hipStream_t stream) {
  (void)in_sizes; (void)n_in; (void)out_size; (void)ws_size;
  const float* x      = (const float*)d_in[0];
  const float* fc1_w  = (const float*)d_in[1];
  const float* fc1_b  = (const float*)d_in[2];
  const float* clstok = (const float*)d_in[3];
  const float* ln1_g  = (const float*)d_in[4];
  const float* ln1_b  = (const float*)d_in[5];
  const float* qkv1_w = (const float*)d_in[6];
  const float* out1_w = (const float*)d_in[7];
  const float* out1_b = (const float*)d_in[8];
  const float* conv1w = (const float*)d_in[9];
  const float* ln2_g  = (const float*)d_in[10];
  const float* ln2_b  = (const float*)d_in[11];
  const float* qkv2_w = (const float*)d_in[12];
  const float* out2_w = (const float*)d_in[13];
  const float* out2_b = (const float*)d_in[14];
  const float* conv2w = (const float*)d_in[15];
  const float* pg7_w  = (const float*)d_in[16];
  const float* pg7_b  = (const float*)d_in[17];
  const float* pg5_w  = (const float*)d_in[18];
  const float* pg5_b  = (const float*)d_in[19];
  const float* pg3_w  = (const float*)d_in[20];
  const float* pg3_b  = (const float*)d_in[21];
  const float* norm_g = (const float*)d_in[22];
  const float* norm_b = (const float*)d_in[23];
  const float* fc2_w  = (const float*)d_in[24];
  const float* fc2_b  = (const float*)d_in[25];

  float* w = (float*)d_ws;
  size_t off = 0;
  auto alloc = [&](size_t n) { float* p = w + off; off += n; return p; };
  Scratch S;
  S.h    = alloc((size_t)NP * DMODEL);
  S.t0   = alloc((size_t)NP * DMODEL);
  S.t1   = alloc((size_t)NP * DMODEL);
  S.qkv  = alloc((size_t)NP * 3 * DMODEL);
  S.qh   = alloc((size_t)NH * NP * DH);
  S.kh   = alloc((size_t)NH * NP * DH);
  S.vh   = alloc((size_t)NH * NP * DH);
  S.oh   = alloc((size_t)NH * NP * DH);
  S.ql   = alloc((size_t)NH * MLM * DH);
  S.kl   = alloc((size_t)NH * MLM * DH);
  S.a2   = alloc((size_t)NH * MLM * MLM);
  S.z    = alloc((size_t)NH * MLM * MLM);
  S.xz   = alloc((size_t)NH * MLM * MLM);
  S.w1   = alloc((size_t)NH * MLM * MLM);
  S.w2   = alloc((size_t)NH * MLM * MLM);
  S.a3v  = alloc((size_t)NH * MLM * DH);
  S.scal = alloc(64);
  S.rsum = alloc((size_t)NH * MLM);
  S.csum = alloc((size_t)NH * MLM);
  S.big0 = alloc((size_t)NH * NP * MLM);
  S.big1 = alloc((size_t)NH * NP * MLM);

  // 1) h = relu(x @ fc1_w + fc1_b) into rows 1..8192; row0 = cls; dup first 89 rows
  gemm(stream, x, fc1_w, fc1_b, S.h + DMODEL, NTOK, DMODEL, INDIM, 0, 0, 0, 1, false, true);
  finish_embed<<<GRID1(ADD_DUP * DMODEL), 256, 0, stream>>>(S.h, clstok);

  // 2) h += nystrom_attn(ln1(h))
  run_attn(stream, S.h, S, ln1_g, ln1_b, qkv1_w, out1_w, out1_b, conv1w);

  // 3) PPEG (depthwise 7/5/3 convs over 91x91 image of tokens)
  copy_buf<<<GRID1(DMODEL), 256, 0, stream>>>(S.h, S.t0, DMODEL); // cls row
  ppeg_kernel<<<GRID1((long long)NIMG * DMODEL), 256, 0, stream>>>(
      S.h, S.t0, pg7_w, pg7_b, pg5_w, pg5_b, pg3_w, pg3_b);
  copy_buf<<<GRID1((long long)NTOT * DMODEL), 256, 0, stream>>>(S.t0, S.h, (long long)NTOT * DMODEL);

  // 4) h += nystrom_attn(ln2(h))
  run_attn(stream, S.h, S, ln2_g, ln2_b, qkv2_w, out2_w, out2_b, conv2w);

  // 5) final norm (row 0 only) + fc2 + softmax + argmax -> d_out
  //    layout: [logits(2) | probs(2) | yhat(1) | emb(512)]
  final_head<<<1, 512, 0, stream>>>(S.h, norm_g, norm_b, fc2_w, fc2_b, (float*)d_out);
}
